// TritonMoE_19550691131408
// MI455X (gfx1250) — matrix operand order repaired and checked by reading, not compiled, not run on hardware
//
#include <hip/hip_runtime.h>
#include <hip/hip_bf16.h>
#include <math.h>

typedef __attribute__((ext_vector_type(16))) __bf16 v16bf;
typedef __attribute__((ext_vector_type(8)))  __bf16 v8bf;
typedef __attribute__((ext_vector_type(8)))  float  v8f;

#define T_TOK 4096
#define H_DIM 768
#define E_EXP 8
#define F_DIM 3072
#define LDW1  (E_EXP * F_DIM)   // 24576, row stride of w1 [H, E*F]
#define K_TOP 2
#define BLK   16                // MoE BLOCK_SIZE

// Packed-weight geometry: B tiles are 32(K) x 16(N) bf16, stored as 1KB blobs in
// exact WMMA B-operand order: lane l's v16bf at blob + l*32 bytes.
// W1 packed: tile id = ((e*192 + nf)*24 + ks)   nf: f-col tile (F/16), ks: K/32 over H
// W2 packed: tile id = ((e*48  + nh)*96 + kf)   nh: h-col tile (H/16), kf: K/32 over F
#define W1_TILES (E_EXP * 192 * 24)               // 36864
#define W2_TILES (E_EXP * 48 * 96)                // 36864
#define TILE_ELEMS 512                            // bf16 per tile (1KB)

// workspace layout (bytes)
#define WS_COEF 0                                   // float[T*E]
#define WS_SEL  (T_TOK * E_EXP * 4)                 // int[T*K]
#define WS_CNT  (WS_SEL + T_TOK * K_TOP * 4)        // int[E]
#define WS_LAST (WS_CNT + E_EXP * 4)                // int[E]
#define WS_PAD  (WS_LAST + E_EXP * 4)               // int[E]
#define WS_W1P  (256 * 1024)                        // bf16[W1_TILES*512] = 36 MiB
#define WS_W2P  (WS_W1P + W1_TILES * TILE_ELEMS * 2)

// ---------------- router / coef pipeline ----------------

__global__ void moe_init(float* coef, int* cnt, int* last) {
    int i = blockIdx.x * 256 + threadIdx.x;
    if (i < T_TOK * E_EXP) coef[i] = 0.0f;
    if (i < E_EXP) { cnt[i] = 0; last[i] = -1; }
}

__global__ void moe_router(const float* __restrict__ x, const float* __restrict__ rw,
                           float* __restrict__ logits_out, int* __restrict__ sel,
                           int* __restrict__ cnt, int* __restrict__ last) {
    int t = blockIdx.x * blockDim.x + threadIdx.x;
    if (t >= T_TOK) return;
    const float* xr = x + (size_t)t * H_DIM;
    float lg[E_EXP];
#pragma unroll
    for (int e = 0; e < E_EXP; ++e) lg[e] = 0.0f;
    for (int h = 0; h < H_DIM; ++h) {
        float xv = xr[h];
#pragma unroll
        for (int e = 0; e < E_EXP; ++e) lg[e] += xv * rw[e * H_DIM + h];
    }
#pragma unroll
    for (int e = 0; e < E_EXP; ++e) logits_out[(size_t)t * E_EXP + e] = lg[e];
    // top-2 on logits (same order as softmax probs); lowest index wins ties
    int i0 = 0;
#pragma unroll
    for (int e = 1; e < E_EXP; ++e) if (lg[e] > lg[i0]) i0 = e;
    int i1 = -1;
#pragma unroll
    for (int e = 0; e < E_EXP; ++e)
        if (e != i0 && (i1 < 0 || lg[e] > lg[i1])) i1 = e;
    sel[t * K_TOP + 0] = i0;
    sel[t * K_TOP + 1] = i1;
    atomicAdd(&cnt[i0], 1);
    atomicAdd(&cnt[i1], 1);
    atomicMax(&last[i0], t * K_TOP + 0);
    atomicMax(&last[i1], t * K_TOP + 1);
}

__global__ void moe_pad(const int* __restrict__ cnt, int* __restrict__ pad) {
    int e = threadIdx.x;
    if (e < E_EXP) pad[e] = (BLK - (cnt[e] % BLK)) % BLK;
}

__global__ void moe_coef(const int* __restrict__ sel, const int* __restrict__ last,
                         const int* __restrict__ pad, float* __restrict__ coef) {
    int i = blockIdx.x * blockDim.x + threadIdx.x;
    if (i >= T_TOK * K_TOP) return;
    int e = sel[i];
    int t = i / K_TOP;
    float m = 1.0f + ((i == last[e]) ? (float)pad[e] : 0.0f);
    atomicAdd(&coef[(size_t)t * E_EXP + e], m);
}

// ---------------- weight packing: f32 row-major -> bf16 WMMA-B-swizzled ----------------

__global__ void pack_w1(const float* __restrict__ w1, __bf16* __restrict__ dst) {
    size_t idx = (size_t)blockIdx.x * 256 + threadIdx.x;
    if (idx >= (size_t)W1_TILES * TILE_ELEMS) return;
    int i    = (int)(idx & 15);
    int l    = (int)((idx >> 4) & 31);
    int tile = (int)(idx >> 9);
    int ks   = tile % 24;   int tmp = tile / 24;
    int nf   = tmp % 192;   int e   = tmp / 192;
    int col  = l & 15;
    int kbof = (l >> 4) * 8;
    int krow = ks * 32 + kbof + (i >> 3) * 16 + (i & 7);
    dst[idx] = (__bf16)w1[(size_t)krow * LDW1 + (size_t)e * F_DIM + nf * 16 + col];
}

__global__ void pack_w2(const float* __restrict__ w2, __bf16* __restrict__ dst) {
    size_t idx = (size_t)blockIdx.x * 256 + threadIdx.x;
    if (idx >= (size_t)W2_TILES * TILE_ELEMS) return;
    int i    = (int)(idx & 15);
    int l    = (int)((idx >> 4) & 31);
    int tile = (int)(idx >> 9);
    int kf   = tile % 96;   int tmp = tile / 96;
    int nh   = tmp % 48;    int e   = tmp / 48;
    int col  = l & 15;
    int kbof = (l >> 4) * 8;
    int frow = e * F_DIM + kf * 32 + kbof + (i >> 3) * 16 + (i & 7);
    dst[idx] = (__bf16)w2[(size_t)frow * H_DIM + nh * 16 + col];
}

// ---------------- fused dense MoE FFN (WMMA bf16, packed weights) ----------------
// 256 blocks x 256 threads (8 wave32). One 16-token M-tile per block.
// Per expert, loop F in chunks of 128:
//   stage A: wave w computes h-tile (chunk*128 + w*16 .. +15) with 24 WMMA over K=768,
//            gelu + coef(row) scale, pack bf16 -> LDS.
//   stage B: wave w accumulates its 6 output N-tiles (cols w*96..) with 24 WMMA/chunk.
// B operands: single 32B contiguous load per WMMA from packed weights (L2-resident).
__global__ __launch_bounds__(256) void moe_ffn(const float* __restrict__ x,
                                               const __bf16* __restrict__ w1p,
                                               const __bf16* __restrict__ w2p,
                                               const float* __restrict__ coef,
                                               float* __restrict__ out) {
    __shared__ __align__(16) __bf16 xs[16][776];  // x tile (16B-aligned rows)
    __shared__ __align__(16) __bf16 hs[16][136];  // h chunk [16 x 128], padded

    const int tid  = threadIdx.x;
    const int wv   = tid >> 5;
    const int lane = tid & 31;
    const int ln16 = lane & 15;
    const int hi8  = (lane >> 4) << 3;      // C-layout row offset: 0 or 8
    const int kaof = (lane >= 16) ? 8 : 0;  // A-operand K offset (16-bit layout)
    const int m0   = blockIdx.x * 16;

    for (int i = tid; i < 16 * H_DIM; i += 256) {
        int r = i / H_DIM, c = i % H_DIM;
        xs[r][c] = (__bf16)x[(size_t)(m0 + r) * H_DIM + c];
    }
    __syncthreads();

    v8f oacc[6];
#pragma unroll
    for (int n = 0; n < 6; ++n) oacc[n] = (v8f){0.f, 0.f, 0.f, 0.f, 0.f, 0.f, 0.f, 0.f};

    for (int e = 0; e < E_EXP; ++e) {
        float cf[8];
#pragma unroll
        for (int r = 0; r < 8; ++r)
            cf[r] = coef[(size_t)(m0 + r + hi8) * E_EXP + e];

        for (int chunk = 0; chunk < 24; ++chunk) {
            // -------- stage A: h = gelu(x @ W1_e[:, chunk]) * coef(row) --------
            {
                const int nf = chunk * 8 + wv;                  // this wave's f-tile
                const __bf16* bt = w1p + ((size_t)(e * 192 + nf) * 24) * TILE_ELEMS
                                       + lane * 16;
                // prefetch next chunk's tile stream (global_prefetch_b8)
                __builtin_prefetch((const char*)(bt + 24 * TILE_ELEMS * 8), 0, 1);
                v8f c = (v8f){0.f, 0.f, 0.f, 0.f, 0.f, 0.f, 0.f, 0.f};
                for (int ks = 0; ks < 24; ++ks) {
                    const int ka = ks * 32 + kaof;
                    v8bf alo = *(const v8bf*)&xs[ln16][ka];
                    v8bf ahi = *(const v8bf*)&xs[ln16][ka + 16];
                    v16bf a = __builtin_shufflevector(alo, ahi,
                        0, 1, 2, 3, 4, 5, 6, 7, 8, 9, 10, 11, 12, 13, 14, 15);
                    v16bf b = *(const v16bf*)(bt + (size_t)ks * TILE_ELEMS);
                    c = __builtin_amdgcn_wmma_f32_16x16x32_bf16(
                        false, a, false, b, (short)0, c, false, false);
                }
#pragma unroll
                for (int r = 0; r < 8; ++r) {
                    float v = c[r];
                    float g = 0.5f * v * (1.0f + erff(v * 0.70710678118654752f));
                    hs[r + hi8][wv * 16 + ln16] = (__bf16)(g * cf[r]);
                }
            }
            __syncthreads();
            // -------- stage B: oacc += h_chunk @ W2_e[chunk rows, wave cols] --------
            {
#pragma unroll
                for (int ks = 0; ks < 4; ++ks) {
                    const int kf = chunk * 4 + ks;
                    const int ka = ks * 32 + kaof;
                    v8bf alo = *(const v8bf*)&hs[ln16][ka];
                    v8bf ahi = *(const v8bf*)&hs[ln16][ka + 16];
                    v16bf a = __builtin_shufflevector(alo, ahi,
                        0, 1, 2, 3, 4, 5, 6, 7, 8, 9, 10, 11, 12, 13, 14, 15);
#pragma unroll
                    for (int n = 0; n < 6; ++n) {
                        const int nh = wv * 6 + n;
                        v16bf b = *(const v16bf*)(w2p
                            + ((size_t)(e * 48 + nh) * 96 + kf) * TILE_ELEMS
                            + lane * 16);
                        oacc[n] = __builtin_amdgcn_wmma_f32_16x16x32_bf16(
                            false, a, false, b, (short)0, oacc[n], false, false);
                    }
                }
            }
            __syncthreads();
        }
    }

#pragma unroll
    for (int n = 0; n < 6; ++n)
#pragma unroll
        for (int r = 0; r < 8; ++r)
            out[(size_t)(m0 + r + hi8) * H_DIM + wv * 96 + n * 16 + ln16] = oacc[n][r];
}

// ---------------- launch ----------------

extern "C" void kernel_launch(void* const* d_in, const int* in_sizes, int n_in,
                              void* d_out, int out_size, void* d_ws, size_t ws_size,
                              hipStream_t stream) {
    const float* x  = (const float*)d_in[0];
    const float* rw = (const float*)d_in[1];
    const float* w1 = (const float*)d_in[2];
    const float* w2 = (const float*)d_in[3];

    float* out    = (float*)d_out;
    float* logits = out + (size_t)T_TOK * H_DIM;   // second tuple output

    char*   ws   = (char*)d_ws;
    float*  coef = (float*)(ws + WS_COEF);
    int*    sel  = (int*)(ws + WS_SEL);
    int*    cnt  = (int*)(ws + WS_CNT);
    int*    last = (int*)(ws + WS_LAST);
    int*    pad  = (int*)(ws + WS_PAD);
    __bf16* w1pk = (__bf16*)(ws + WS_W1P);
    __bf16* w2pk = (__bf16*)(ws + WS_W2P);

    moe_init<<<(T_TOK * E_EXP + 255) / 256, 256, 0, stream>>>(coef, cnt, last);
    moe_router<<<(T_TOK + 255) / 256, 256, 0, stream>>>(x, rw, logits, sel, cnt, last);
    moe_pad<<<1, 32, 0, stream>>>(cnt, pad);
    moe_coef<<<(T_TOK * K_TOP + 255) / 256, 256, 0, stream>>>(sel, last, pad, coef);

    const int pk_grid = (int)(((size_t)W1_TILES * TILE_ELEMS + 255) / 256);
    pack_w1<<<pk_grid, 256, 0, stream>>>(w1, w1pk);
    pack_w2<<<pk_grid, 256, 0, stream>>>(w2, w2pk);

    moe_ffn<<<T_TOK / 16, 256, 0, stream>>>(x, w1pk, w2pk, coef, out);
}
